// QuarotAttention_48919677501574
// MI455X (gfx1250) — compile-verified, matrix operand order checked by reading, not run on hardware
//
#include <hip/hip_runtime.h>

// ---------------------------------------------------------------------------
// QuaRot attention for gfx1250 (MI455X): bf16 WMMA everywhere, flash softmax.
// Round 3: flash kernel stages the K chunk in LDS (once per block instead of
// once per wave, 8x less L2->WGP traffic); K tile shares its LDS region with
// the P scratch (never simultaneously live), keeping LDS at 56.6 KB.
// ---------------------------------------------------------------------------

typedef __bf16 bf16_t;
typedef __attribute__((ext_vector_type(16))) __bf16 v16bf;
typedef __attribute__((ext_vector_type(8)))  __bf16 v8bf;
typedef __attribute__((ext_vector_type(8)))  float  v8f;

#define DIM     1152
#define NHEAD   16
#define HDIM    72
#define DPQ     96      // head dim padded for K=32 wmma steps (3 steps)
#define DPV     80      // head dim padded for 16-wide output tiles (5 tiles)
#define SEQ     2048
#define BATCH   2
#define MROWS   (BATCH*SEQ)   // 4096

union FragU { v16bf v; v8bf h[2]; };

// ---------------------------------------------------------------------------
// Prep kernels
// ---------------------------------------------------------------------------

__global__ __launch_bounds__(256) void fold_rot_kernel(
    const float* __restrict__ W, const float* __restrict__ Qrot,
    bf16_t* __restrict__ Wout)
{
    int idx = blockIdx.x * blockDim.x + threadIdx.x;
    if (idx >= DIM * DIM) return;
    int row = idx / DIM, col = idx % DIM;
    int h = col / HDIM, d = col % HDIM;
    const float* wrow = W + (size_t)row * DIM + h * HDIM;
    float s = 0.f;
    #pragma unroll 8
    for (int e = 0; e < HDIM; e++) s += wrow[e] * Qrot[e * HDIM + d];
    Wout[idx] = (bf16_t)s;
}

__global__ void fold_bias_kernel(const float* __restrict__ b,
                                 const float* __restrict__ Qrot,
                                 float* __restrict__ bout)
{
    int col = blockIdx.x * blockDim.x + threadIdx.x;
    if (col >= DIM) return;
    int h = col / HDIM, d = col % HDIM;
    float s = 0.f;
    for (int e = 0; e < HDIM; e++) s += b[h * HDIM + e] * Qrot[e * HDIM + d];
    bout[col] = s;
}

__global__ void cvt_f32_bf16_kernel(const float* __restrict__ src,
                                    bf16_t* __restrict__ dst, int n)
{
    int i = blockIdx.x * blockDim.x + threadIdx.x;
    if (i < n) dst[i] = (bf16_t)src[i];
}

__global__ void zero_u32_kernel(unsigned int* __restrict__ p, int n)
{
    int i = blockIdx.x * blockDim.x + threadIdx.x;
    if (i < n) p[i] = 0u;
}

// ---------------------------------------------------------------------------
// bf16 GEMM, fp32 accumulate, WMMA 16x16x32.  C = A[4096,1152]@B[1152,1152]+bias
// MODE 0: fp32 row-major out (final projection)
// MODE 1: bf16 head-major padded [BATCH, NHEAD, SEQ, DP] out, * scale
// 128x128 tile / WG, 8 waves (16 rows x 128 cols each), K step 32,
// register-staged double-buffered LDS, pipelined B-fragment loads.
// ---------------------------------------------------------------------------
template<int MODE, int DP>
__global__ __launch_bounds__(256) void gemm_bf16_kernel(
    const bf16_t* __restrict__ A, const bf16_t* __restrict__ Bm,
    const float* __restrict__ bias, void* __restrict__ outp, float scale)
{
    constexpr int K = DIM, NC = DIM;

    __shared__ bf16_t As [2][128][40];   // [buf][row][k], rows 16B-aligned
    __shared__ bf16_t BsT[2][128][40];   // [buf][col][k] (B tile transposed)

    const int tid  = threadIdx.x;
    const int lane = tid & 31;
    const int w    = tid >> 5;
    const int ncol = lane & 15;
    const int hh   = lane >> 4;
    const int m0   = blockIdx.y * 128;
    const int n0   = blockIdx.x * 128;

    // global-load staging (per-thread registers)
    const int arow = tid >> 1, ach = tid & 1;         // A: 128 rows x 2 chunks
    const int bkk  = tid & 31, bnc = tid >> 5;        // B: 32 k-rows x 8 chunks
    uint4 ra0, ra1;
    bf16_t rb[16] __attribute__((aligned(16)));

    auto gload = [&](int k0) {
        const uint4* srcA = (const uint4*)(A + (size_t)(m0 + arow) * K + k0 + ach * 16);
        ra0 = srcA[0];
        ra1 = srcA[1];
        const bf16_t* srcB = Bm + (size_t)(k0 + bkk) * NC + n0 + bnc * 16;
        *(uint4*)&rb[0] = *(const uint4*)srcB;
        *(uint4*)&rb[8] = *(const uint4*)(srcB + 8);
    };
    auto lstore = [&](int buf) {
        *(uint4*)&As[buf][arow][ach * 16]     = ra0;
        *(uint4*)&As[buf][arow][ach * 16 + 8] = ra1;
        #pragma unroll
        for (int i = 0; i < 16; i++) BsT[buf][bnc * 16 + i][bkk] = rb[i];
    };

    v8f zero;
    #pragma unroll
    for (int e = 0; e < 8; e++) zero[e] = 0.f;
    v8f acc[8];
    #pragma unroll
    for (int nt = 0; nt < 8; nt++) acc[nt] = zero;

    const int mrow = w * 16 + ncol;

    gload(0);
    lstore(0);

    constexpr int NSTEP = K / 32;  // 36
    for (int step = 0; step < NSTEP; step++) {
        const int cur = step & 1;
        __syncthreads();
        if (step + 1 < NSTEP) gload((step + 1) * 32);   // prefetch next tile -> regs

        // --- compute from LDS[cur], 1-deep pipelined B fragments ---
        FragU af;
        af.h[0] = *(const v8bf*)&As[cur][mrow][hh * 8];
        af.h[1] = *(const v8bf*)&As[cur][mrow][16 + hh * 8];
        FragU bc;
        bc.h[0] = *(const v8bf*)&BsT[cur][ncol][hh * 8];
        bc.h[1] = *(const v8bf*)&BsT[cur][ncol][16 + hh * 8];
        #pragma unroll
        for (int nt = 0; nt < 8; nt++) {
            FragU bn;
            if (nt < 7) {
                bn.h[0] = *(const v8bf*)&BsT[cur][(nt + 1) * 16 + ncol][hh * 8];
                bn.h[1] = *(const v8bf*)&BsT[cur][(nt + 1) * 16 + ncol][16 + hh * 8];
            } else {
                bn = bc;
            }
            acc[nt] = __builtin_amdgcn_wmma_f32_16x16x32_bf16(
                false, af.v, false, bc.v, (short)0, acc[nt], false, false);
            bc = bn;
        }

        if (step + 1 < NSTEP) lstore((step + 1) & 1);   // drain regs -> other buffer
    }

    // --- epilogue (C layout: col = lane%16, row = vgpr + 8*half) ---
    const int mbase = m0 + w * 16;
    #pragma unroll
    for (int nt = 0; nt < 8; nt++) {
        const int col = n0 + nt * 16 + ncol;
        const float bcol = bias[col];
        if (MODE == 0) {
            float* op = (float*)outp;
            #pragma unroll
            for (int r = 0; r < 8; r++) {
                int row = mbase + r + 8 * hh;
                op[(size_t)row * NC + col] = acc[nt][r] + bcol;
            }
        } else {
            const int head = col / HDIM;     // compile-time magic (HDIM const)
            const int d    = col % HDIM;
            bf16_t* op = (bf16_t*)outp;
            #pragma unroll
            for (int r = 0; r < 8; r++) {
                int row = mbase + r + 8 * hh;
                int bi  = row >> 11;          // / SEQ
                int nr  = row & (SEQ - 1);
                size_t off = ((size_t)(bi * NHEAD + head) * SEQ + nr) * DP + d;
                op[off] = (bf16_t)((acc[nt][r] + bcol) * scale);
            }
        }
    }
}

// ---------------------------------------------------------------------------
// Flash attention: per (b,h), 128-query tile per WG, 8 waves x 16 rows.
// Q/K in [BH, SEQ, 96] bf16 (1/sqrt(D) folded into Q), V in [BH, SEQ, 80].
// K chunk and V chunk staged in LDS once per block; the K tile shares its LDS
// region with the per-wave P scratch (disjoint lifetimes within a chunk).
// ---------------------------------------------------------------------------
union KsPlU {
    bf16_t ks[128][104];      // K chunk, row-major [key][kdim], 26.6 KB
    bf16_t pl[8][16][136];    // per-wave probabilities, 34.8 KB
};

__global__ __launch_bounds__(256) void flash_attn_kernel(
    const bf16_t* __restrict__ Qb, const bf16_t* __restrict__ Kb,
    const bf16_t* __restrict__ Vb, bf16_t* __restrict__ AO)
{
    __shared__ KsPlU  U;                 // K tile / P scratch (aliased)
    __shared__ bf16_t Vt[DPV][136];      // V chunk transposed: [d][key]

    const int tid  = threadIdx.x;
    const int lane = tid & 31;
    const int w    = tid >> 5;
    const int ncol = lane & 15;
    const int hh   = lane >> 4;
    const int bh   = blockIdx.y;
    const int bi   = bh >> 4;
    const int head = bh & 15;
    const int q0   = blockIdx.x * 128;

    const size_t baseQK = (size_t)bh * SEQ * DPQ;
    const size_t baseV  = (size_t)bh * SEQ * DPV;

    // Q fragments for this wave's 16 rows, loaded once (3 k-steps over 96)
    FragU qf[3];
    {
        const bf16_t* qrow = Qb + baseQK + (size_t)(q0 + w * 16 + ncol) * DPQ;
        #pragma unroll
        for (int ks = 0; ks < 3; ks++) {
            qf[ks].h[0] = *(const v8bf*)(qrow + ks * 32 + hh * 8);
            qf[ks].h[1] = *(const v8bf*)(qrow + ks * 32 + 16 + hh * 8);
        }
    }

    v8f zero;
    #pragma unroll
    for (int e = 0; e < 8; e++) zero[e] = 0.f;

    float mrow[8], lrow[8];
    v8f accd[5];
    #pragma unroll
    for (int r = 0; r < 8; r++) { mrow[r] = -3.0e38f; lrow[r] = 0.f; }
    #pragma unroll
    for (int dt = 0; dt < 5; dt++) accd[dt] = zero;

    for (int kc = 0; kc < SEQ; kc += 128) {
        // barrier 1: previous chunk's PV reads of Vt and U.pl are complete
        __syncthreads();

        // --- stage V chunk transposed: [key][d] -> Vt[d][key] ---
        {
            const int key  = tid >> 1;
            const int half = tid & 1;
            const bf16_t* src = Vb + baseV + (size_t)(kc + key) * DPV + half * 40;
            bf16_t tmp[40] __attribute__((aligned(16)));
            #pragma unroll
            for (int i = 0; i < 5; i++)
                *(uint4*)&tmp[i * 8] = *(const uint4*)(src + i * 8);
            #pragma unroll
            for (int i = 0; i < 40; i++) Vt[half * 40 + i][key] = tmp[i];
        }
        // --- stage K chunk row-major into U.ks (once per block, not per wave) ---
        {
            const int key  = tid >> 1;
            const int half = tid & 1;     // 24 elems each, at offset half*48
            const bf16_t* src = Kb + baseQK + (size_t)(kc + key) * DPQ + half * 48;
            #pragma unroll
            for (int i = 0; i < 3; i++)
                *(uint4*)&U.ks[key][half * 48 + i * 8] = *(const uint4*)(src + i * 8);
        }
        // barrier 2: staging complete
        __syncthreads();

        // --- S = Q @ K^T from LDS (ks outer; 1-deep pipelined B fragments) ---
        v8f s[8];
        #pragma unroll
        for (int nt = 0; nt < 8; nt++) s[nt] = zero;
        #pragma unroll
        for (int ks = 0; ks < 3; ks++) {
            FragU kcur;
            kcur.h[0] = *(const v8bf*)&U.ks[ncol][ks * 32 + hh * 8];
            kcur.h[1] = *(const v8bf*)&U.ks[ncol][ks * 32 + 16 + hh * 8];
            #pragma unroll
            for (int nt = 0; nt < 8; nt++) {
                FragU kn;
                if (nt < 7) {
                    kn.h[0] = *(const v8bf*)&U.ks[(nt + 1) * 16 + ncol][ks * 32 + hh * 8];
                    kn.h[1] = *(const v8bf*)&U.ks[(nt + 1) * 16 + ncol][ks * 32 + 16 + hh * 8];
                } else {
                    kn = kcur;
                }
                s[nt] = __builtin_amdgcn_wmma_f32_16x16x32_bf16(
                    false, qf[ks].v, false, kcur.v, (short)0, s[nt], false, false);
                kcur = kn;
            }
        }
        // barrier 3: all waves done reading U.ks before U.pl writes clobber it
        __syncthreads();

        // --- online softmax (row = r + 8*hh; reductions stay in lane half) ---
        float mnew[8];
        #pragma unroll
        for (int r = 0; r < 8; r++) {
            float mx = s[0][r];
            #pragma unroll
            for (int nt = 1; nt < 8; nt++) mx = fmaxf(mx, s[nt][r]);
            mx = fmaxf(mx, __shfl_xor(mx, 1));
            mx = fmaxf(mx, __shfl_xor(mx, 2));
            mx = fmaxf(mx, __shfl_xor(mx, 4));
            mx = fmaxf(mx, __shfl_xor(mx, 8));
            mnew[r] = fmaxf(mrow[r], mx);
            float alpha = __expf(mrow[r] - mnew[r]);
            mrow[r] = mnew[r];

            float rs = 0.f;
            #pragma unroll
            for (int nt = 0; nt < 8; nt++) {
                float p = __expf(s[nt][r] - mnew[r]);
                rs += p;
                U.pl[w][r + 8 * hh][nt * 16 + ncol] = (bf16_t)p;  // wave-private slice
            }
            rs += __shfl_xor(rs, 1);
            rs += __shfl_xor(rs, 2);
            rs += __shfl_xor(rs, 4);
            rs += __shfl_xor(rs, 8);
            lrow[r] = lrow[r] * alpha + rs;
            #pragma unroll
            for (int dt = 0; dt < 5; dt++) accd[dt][r] *= alpha;
        }

        // --- accd += P @ V (4 k-steps over 128 keys, 5 d tiles, pipelined) ---
        #pragma unroll
        for (int ks2 = 0; ks2 < 4; ks2++) {
            FragU pf;
            pf.h[0] = *(const v8bf*)&U.pl[w][ncol][ks2 * 32 + hh * 8];
            pf.h[1] = *(const v8bf*)&U.pl[w][ncol][ks2 * 32 + 16 + hh * 8];
            FragU vc;
            vc.h[0] = *(const v8bf*)&Vt[ncol][ks2 * 32 + hh * 8];
            vc.h[1] = *(const v8bf*)&Vt[ncol][ks2 * 32 + 16 + hh * 8];
            #pragma unroll
            for (int dt = 0; dt < 5; dt++) {
                FragU vn;
                if (dt < 4) {
                    vn.h[0] = *(const v8bf*)&Vt[(dt + 1) * 16 + ncol][ks2 * 32 + hh * 8];
                    vn.h[1] = *(const v8bf*)&Vt[(dt + 1) * 16 + ncol][ks2 * 32 + 16 + hh * 8];
                } else {
                    vn = vc;
                }
                accd[dt] = __builtin_amdgcn_wmma_f32_16x16x32_bf16(
                    false, pf.v, false, vc.v, (short)0, accd[dt], false, false);
                vc = vn;
            }
        }
    }

    // --- finalize: out = acc / l, scatter into [B*N, 1152] bf16 ---
    #pragma unroll
    for (int r = 0; r < 8; r++) {
        float inv = 1.0f / lrow[r];
        int qrow = q0 + w * 16 + r + 8 * hh;
        size_t rowoff = ((size_t)bi * SEQ + qrow) * DIM + head * HDIM;
        #pragma unroll
        for (int dt = 0; dt < 5; dt++) {
            int d = dt * 16 + ncol;
            if (d < HDIM) AO[rowoff + d] = (bf16_t)(accd[dt][r] * inv);
        }
    }
}

// ---------------------------------------------------------------------------
// Host-side orchestration
// ---------------------------------------------------------------------------
extern "C" void kernel_launch(void* const* d_in, const int* in_sizes, int n_in,
                              void* d_out, int out_size, void* d_ws, size_t ws_size,
                              hipStream_t stream)
{
    (void)in_sizes; (void)n_in; (void)out_size; (void)ws_size;

    const float* x    = (const float*)d_in[0];
    const float* Wq   = (const float*)d_in[1];
    const float* bq   = (const float*)d_in[2];
    const float* Wk   = (const float*)d_in[3];
    const float* bk   = (const float*)d_in[4];
    const float* Wv   = (const float*)d_in[5];
    const float* bv   = (const float*)d_in[6];
    const float* Qrot = (const float*)d_in[7];
    const float* Wp   = (const float*)d_in[8];
    const float* bp   = (const float*)d_in[9];
    float* out = (float*)d_out;

    char* ws = (char*)d_ws;
    size_t off = 0;
    auto carve = [&](size_t bytes) -> char* {
        char* p = ws + off;
        off = (off + bytes + 255) & ~(size_t)255;
        return p;
    };
    bf16_t* xb   = (bf16_t*)carve((size_t)MROWS * DIM * 2);
    bf16_t* wqr  = (bf16_t*)carve((size_t)DIM * DIM * 2);
    bf16_t* wkr  = (bf16_t*)carve((size_t)DIM * DIM * 2);
    bf16_t* wvb  = (bf16_t*)carve((size_t)DIM * DIM * 2);
    bf16_t* wpb  = (bf16_t*)carve((size_t)DIM * DIM * 2);
    float*  bqr  = (float*) carve((size_t)DIM * 4);
    float*  bkr  = (float*) carve((size_t)DIM * 4);
    bf16_t* qbuf = (bf16_t*)carve((size_t)BATCH * NHEAD * SEQ * DPQ * 2);
    bf16_t* kbuf = (bf16_t*)carve((size_t)BATCH * NHEAD * SEQ * DPQ * 2);
    bf16_t* vbuf = (bf16_t*)carve((size_t)BATCH * NHEAD * SEQ * DPV * 2);
    bf16_t* aob  = (bf16_t*)carve((size_t)MROWS * DIM * 2);

    const int nthr = 256;
    auto blocks = [](int n) { return (n + 255) / 256; };

    cvt_f32_bf16_kernel<<<blocks(MROWS * DIM), nthr, 0, stream>>>(x, xb, MROWS * DIM);
    cvt_f32_bf16_kernel<<<blocks(DIM * DIM),   nthr, 0, stream>>>(Wv, wvb, DIM * DIM);
    cvt_f32_bf16_kernel<<<blocks(DIM * DIM),   nthr, 0, stream>>>(Wp, wpb, DIM * DIM);
    fold_rot_kernel<<<blocks(DIM * DIM), nthr, 0, stream>>>(Wq, Qrot, wqr);
    fold_rot_kernel<<<blocks(DIM * DIM), nthr, 0, stream>>>(Wk, Qrot, wkr);
    fold_bias_kernel<<<blocks(DIM), nthr, 0, stream>>>(bq, Qrot, bqr);
    fold_bias_kernel<<<blocks(DIM), nthr, 0, stream>>>(bk, Qrot, bkr);
    {
        int nq = BATCH * NHEAD * SEQ * DPQ / 2;   // u32 count
        int nv = BATCH * NHEAD * SEQ * DPV / 2;
        zero_u32_kernel<<<blocks(nq), nthr, 0, stream>>>((unsigned int*)qbuf, nq);
        zero_u32_kernel<<<blocks(nq), nthr, 0, stream>>>((unsigned int*)kbuf, nq);
        zero_u32_kernel<<<blocks(nv), nthr, 0, stream>>>((unsigned int*)vbuf, nv);
    }

    dim3 ggrid(DIM / 128, MROWS / 128);
    const float qscale = 0.11785113019775793f;  // 72^-0.5
    gemm_bf16_kernel<1, DPQ><<<ggrid, nthr, 0, stream>>>(xb, wqr, bqr, qbuf, qscale);
    gemm_bf16_kernel<1, DPQ><<<ggrid, nthr, 0, stream>>>(xb, wkr, bkr, kbuf, 1.0f);
    gemm_bf16_kernel<1, DPV><<<ggrid, nthr, 0, stream>>>(xb, wvb, bv,  vbuf, 1.0f);

    dim3 fgrid(SEQ / 128, BATCH * NHEAD);
    flash_attn_kernel<<<fgrid, nthr, 0, stream>>>(qbuf, kbuf, vbuf, aob);

    gemm_bf16_kernel<0, 1><<<ggrid, nthr, 0, stream>>>(aob, wpb, bp, out, 1.0f);
}
